// MultiAttnLayer_47579647705165
// MI455X (gfx1250) — matrix operand order repaired and checked by reading, not compiled, not run on hardware
//
#include <hip/hip_runtime.h>
#include <hip/hip_bf16.h>

typedef _Float16 v4h  __attribute__((ext_vector_type(4)));
typedef _Float16 v8h  __attribute__((ext_vector_type(8)));
typedef _Float16 v16h __attribute__((ext_vector_type(16)));
typedef float    v8f  __attribute__((ext_vector_type(8)));

// Async global->LDS copy path (CDNA5 GLOBAL_LOAD_ASYNC_TO_LDS_B128, ASYNCcnt).
// Guarded: falls back to VGPR round-trip when builtins are not declared.
#if defined(__gfx1250__) && __has_builtin(__builtin_amdgcn_global_load_async_to_lds_b128) && __has_builtin(__builtin_amdgcn_s_wait_asynccnt)
#define USE_ASYNC_LDS 1
typedef int v4i_t __attribute__((ext_vector_type(4)));
typedef __attribute__((address_space(1))) v4i_t* as1_v4i;   // global int4*
typedef __attribute__((address_space(3))) v4i_t* as3_v4i;   // LDS int4*
#else
#define USE_ASYNC_LDS 0
#endif

__device__ __forceinline__ v8f vzero8() {
    v8f z;
#pragma unroll
    for (int i = 0; i < 8; ++i) z[i] = 0.0f;
    return z;
}

__device__ __forceinline__ v16h mk16(v8h lo, v8h hi) {
    v16h r;
#pragma unroll
    for (int i = 0; i < 8; ++i) { r[i] = lo[i]; r[i + 8] = hi[i]; }
    return r;
}

// ---------------------------------------------------------------------------
// Tiled GEMM:  C[M,N] = A[M,K] * W[N,K]^T (+bias, +opt relu, +opt residual)
// Block tile 128x128, 8 waves (2x4), each wave 64x32 (4x2 WMMA tiles).
// BK=64, double-buffered dynamic LDS (73728 B). Fragment layouts follow
// CDNA5 ISA 7.12.2. All fragment loads batched ahead of the WMMA chain.
// ---------------------------------------------------------------------------
#define GEMM_SMEM_BYTES (4 * 128 * 72 * 2)   // sA[2] + sB[2], f16, LDA=72

template<bool A_F16, bool RELU, bool RESID, bool OUT32, bool OUT16>
__global__ __launch_bounds__(256)
void gemm_kernel(const void* __restrict__ Ap, const float* __restrict__ Wp,
                 const float* __restrict__ bias, const float* __restrict__ resid,
                 float* __restrict__ out32, _Float16* __restrict__ out16,
                 int M, int N, int K)
{
    constexpr int LDA = 72;                 // f16 elems; 144B row stride (16B mult)
    extern __shared__ char gsmem[];
    _Float16* sA = (_Float16*)gsmem;        // 2 buffers of 128*72
    _Float16* sB = sA + 2 * 128 * LDA;      // 2 buffers of 128*72

    const int t    = threadIdx.x;
    const int lane = t & 31;
    const int wid  = t >> 5;
    const int wm   = (wid >> 2) * 64;       // wave row offset (0/64)
    const int wn   = (wid & 3) * 32;        // wave col offset (0..96)
    const int rowBase = blockIdx.y * 128;
    const int colBase = blockIdx.x * 128;

    const float*    A32 = (const float*)Ap;
    const _Float16* A16 = (const _Float16*)Ap;

    v8f acc[4][2];
#pragma unroll
    for (int mi = 0; mi < 4; ++mi)
#pragma unroll
        for (int ni = 0; ni < 2; ++ni) acc[mi][ni] = vzero8();

    auto loadTile = [&](int buf, int k0) {
        _Float16* dA = sA + buf * 128 * LDA;
        _Float16* dB = sB + buf * 128 * LDA;
        if (A_F16) {
#pragma unroll
            for (int i = 0; i < 4; ++i) {
                int slot = t + i * 256;                 // 1024 slots of 8 f16 (16B)
                int r = slot >> 3, c = (slot & 7) * 8;
                const _Float16* gp = A16 + (size_t)(rowBase + r) * K + k0 + c;
#if USE_ASYNC_LDS
                __builtin_amdgcn_global_load_async_to_lds_b128(
                    (as1_v4i)gp, (as3_v4i)(dA + r * LDA + c), 0, 0);
#else
                *(v8h*)(dA + r * LDA + c) = *(const v8h*)gp;
#endif
            }
        } else {
#pragma unroll
            for (int i = 0; i < 8; ++i) {
                int slot = t + i * 256;                 // 2048 slots of 4 f32
                int r = slot >> 4, c = (slot & 15) * 4;
                float4 f = *(const float4*)(A32 + (size_t)(rowBase + r) * K + k0 + c);
                v4h h; h[0] = (_Float16)f.x; h[1] = (_Float16)f.y;
                       h[2] = (_Float16)f.z; h[3] = (_Float16)f.w;
                *(v4h*)(dA + r * LDA + c) = h;
            }
        }
#pragma unroll
        for (int i = 0; i < 8; ++i) {
            int slot = t + i * 256;
            int r = slot >> 4, c = (slot & 15) * 4;
            float4 f = *(const float4*)(Wp + (size_t)(colBase + r) * K + k0 + c);
            v4h h; h[0] = (_Float16)f.x; h[1] = (_Float16)f.y;
                   h[2] = (_Float16)f.z; h[3] = (_Float16)f.w;
            *(v4h*)(dB + r * LDA + c) = h;
        }
    };

    loadTile(0, 0);
    const int KT = K >> 6;
    const int kh = (lane >> 4) * 8;
    for (int kt = 0; kt < KT; ++kt) {
#if USE_ASYNC_LDS
        if (A_F16) __builtin_amdgcn_s_wait_asynccnt(0);
#endif
        __syncthreads();
        if (kt + 1 < KT) loadTile((kt + 1) & 1, (kt + 1) << 6);
        const _Float16* dA = sA + (kt & 1) * 128 * LDA;
        const _Float16* dB = sB + (kt & 1) * 128 * LDA;

#pragma unroll
        for (int ks = 0; ks < 2; ++ks) {
            const int kk = ks * 32;
            // Batch all fragment loads first, then run the WMMA chain.
            v16h a[4], bf[2];
#pragma unroll
            for (int mi = 0; mi < 4; ++mi) {
                const _Float16* pa = &dA[(wm + mi * 16 + (lane & 15)) * LDA + kk];
                a[mi] = mk16(*(const v8h*)(pa + kh), *(const v8h*)(pa + 16 + kh));
            }
#pragma unroll
            for (int ni = 0; ni < 2; ++ni) {
                const _Float16* pb = &dB[(wn + ni * 16 + (lane & 15)) * LDA + kk + kh];
                bf[ni] = mk16(*(const v8h*)(pb), *(const v8h*)(pb + 16));
            }
#pragma unroll
            for (int ni = 0; ni < 2; ++ni)
#pragma unroll
                for (int mi = 0; mi < 4; ++mi)
                    acc[mi][ni] = __builtin_amdgcn_wmma_f32_16x16x32_f16(
                        false, a[mi], false, bf[ni], (short)0, acc[mi][ni], false, false);
        }
    }

    // Epilogue: C VGPR r -> row r (+8 for lanes 16..31), col = lane&15
#pragma unroll
    for (int ni = 0; ni < 2; ++ni) {
        const int col = colBase + wn + ni * 16 + (lane & 15);
        const float bv = bias[col];
#pragma unroll
        for (int mi = 0; mi < 4; ++mi) {
#pragma unroll
            for (int r = 0; r < 8; ++r) {
                const int row = rowBase + wm + mi * 16 + r + ((lane >> 4) << 3);
                float v = acc[mi][ni][r] + bv;
                if (RELU) v = v > 0.0f ? v : 0.0f;
                const size_t idx = (size_t)row * N + col;
                if (RESID) v += resid[idx];
                if (OUT32) out32[idx] = v;
                if (OUT16) out16[idx] = (_Float16)v;
            }
        }
    }
}

// ---------------------------------------------------------------------------
// Block-local attention: one workgroup per (batch, window, head).
// Q(128x64) K^T -> scores(128x128) via WMMA, fp32 softmax, P(128x128) V -> ctx.
// ---------------------------------------------------------------------------
#define ATTN_SMEM_BYTES 156672

__global__ __launch_bounds__(256)
void attn_kernel(const _Float16* __restrict__ qkv, _Float16* __restrict__ ctxOut)
{
    extern __shared__ char smem[];
    _Float16* sQ  = (_Float16*)smem;              // 128 x 72
    _Float16* sK  = sQ + 128 * 72;                // 128 x 72
    _Float16* sVt = sK + 128 * 72;                // 64  x 136 (V transposed)
    float*    sS  = (float*)(sVt + 64 * 136);     // 128 x 132 fp32 scores
    _Float16* sP  = (_Float16*)((char*)sS + 128 * 132 * 4); // 128 x 136 probs

    const int t = threadIdx.x, lane = t & 31, wv = t >> 5;
    const int head = blockIdx.x & 15;
    const int win  = (blockIdx.x >> 4) & 63;
    const int b    = blockIdx.x >> 10;
    const size_t tok0 = (size_t)b * 8192 + (size_t)win * 128;
    const int kh = (lane >> 4) * 8;

    // Stage Q, K (row-major) and V transposed into LDS, f16.
    {
        const int r  = t >> 1;
        const int c0 = (t & 1) * 32;
        const size_t base = (tok0 + r) * 3072 + head * 64 + c0;
#pragma unroll
        for (int j = 0; j < 4; ++j) {
            v8h q = *(const v8h*)(qkv + base + j * 8);
            *(v8h*)(&sQ[r * 72 + c0 + j * 8]) = q;
            v8h k = *(const v8h*)(qkv + base + 1024 + j * 8);
            *(v8h*)(&sK[r * 72 + c0 + j * 8]) = k;
            v8h v = *(const v8h*)(qkv + base + 2048 + j * 8);
#pragma unroll
            for (int e = 0; e < 8; ++e) sVt[(c0 + j * 8 + e) * 136 + r] = v[e];
        }
    }
    __syncthreads();

    // Scores: wave wv owns Q rows [wv*16, wv*16+16), all 128 key columns.
    {
        const int m0 = wv * 16;
        v8f accs[8];
#pragma unroll
        for (int nt = 0; nt < 8; ++nt) accs[nt] = vzero8();
#pragma unroll
        for (int ks = 0; ks < 2; ++ks) {
            const int kk = ks * 32;
            const _Float16* pa = &sQ[(m0 + (lane & 15)) * 72 + kk];
            v16h a = mk16(*(const v8h*)(pa + kh), *(const v8h*)(pa + 16 + kh));
            v16h bb[8];
#pragma unroll
            for (int nt = 0; nt < 8; ++nt) {
                const _Float16* pb = &sK[(nt * 16 + (lane & 15)) * 72 + kk + kh];
                bb[nt] = mk16(*(const v8h*)pb, *(const v8h*)(pb + 16));
            }
#pragma unroll
            for (int nt = 0; nt < 8; ++nt)
                accs[nt] = __builtin_amdgcn_wmma_f32_16x16x32_f16(
                    false, a, false, bb[nt], (short)0, accs[nt], false, false);
        }
#pragma unroll
        for (int nt = 0; nt < 8; ++nt)
#pragma unroll
            for (int r = 0; r < 8; ++r)
                sS[(m0 + r + ((lane >> 4) << 3)) * 132 + nt * 16 + (lane & 15)] =
                    accs[nt][r] * 0.125f;    // 1/sqrt(64)
    }
    __syncthreads();

    // Row-wise softmax, fp32 -> f16 probs.
    if (t < 128) {
        float* row = &sS[t * 132];
        float mx = row[0];
        for (int j = 1; j < 128; ++j) mx = fmaxf(mx, row[j]);
        float s = 0.0f;
        for (int j = 0; j < 128; ++j) { float e = __expf(row[j] - mx); row[j] = e; s += e; }
        const float inv = 1.0f / s;
        _Float16* prow = &sP[t * 136];
        for (int j = 0; j < 128; ++j) prow[j] = (_Float16)(row[j] * inv);
    }
    __syncthreads();

    // ctx = P(128x128) x V(128x64)  (B fragment from transposed V)
    {
        const int m0 = wv * 16;
        v8f acc2[4];
#pragma unroll
        for (int nt = 0; nt < 4; ++nt) acc2[nt] = vzero8();
#pragma unroll
        for (int ks = 0; ks < 4; ++ks) {
            const int kk = ks * 32;
            const _Float16* pa = &sP[(m0 + (lane & 15)) * 136 + kk];
            v16h a = mk16(*(const v8h*)(pa + kh), *(const v8h*)(pa + 16 + kh));
            v16h bb[4];
#pragma unroll
            for (int nt = 0; nt < 4; ++nt) {
                const _Float16* pb = &sVt[(nt * 16 + (lane & 15)) * 136 + kk + kh];
                bb[nt] = mk16(*(const v8h*)pb, *(const v8h*)(pb + 16));
            }
#pragma unroll
            for (int nt = 0; nt < 4; ++nt)
                acc2[nt] = __builtin_amdgcn_wmma_f32_16x16x32_f16(
                    false, a, false, bb[nt], (short)0, acc2[nt], false, false);
        }
#pragma unroll
        for (int nt = 0; nt < 4; ++nt)
#pragma unroll
            for (int r = 0; r < 8; ++r) {
                const size_t row = tok0 + m0 + r + ((lane >> 4) << 3);
                const int col = head * 64 + nt * 16 + (lane & 15);
                ctxOut[row * 1024 + col] = (_Float16)acc2[nt][r];
            }
    }
}

// ---------------------------------------------------------------------------
// LayerNorm over D=1024: one block per token row.
// ---------------------------------------------------------------------------
template<bool H16>
__global__ __launch_bounds__(256)
void ln_kernel(const float* __restrict__ x, const float* __restrict__ g,
               const float* __restrict__ bt, float* __restrict__ y,
               _Float16* __restrict__ y16)
{
    __shared__ float r1[256], r2[256];
    const int row = blockIdx.x;
    const int tid = threadIdx.x;
    const float* xr = x + (size_t)row * 1024;

    float v[4], s1 = 0.0f, s2 = 0.0f;
#pragma unroll
    for (int i = 0; i < 4; ++i) {
        v[i] = xr[tid + i * 256];
        s1 += v[i]; s2 += v[i] * v[i];
    }
    r1[tid] = s1; r2[tid] = s2;
    __syncthreads();
    for (int s = 128; s > 0; s >>= 1) {
        if (tid < s) { r1[tid] += r1[tid + s]; r2[tid] += r2[tid + s]; }
        __syncthreads();
    }
    const float mu  = r1[0] * (1.0f / 1024.0f);
    const float var = r2[0] * (1.0f / 1024.0f) - mu * mu;
    const float inv = rsqrtf(var + 1e-5f);
#pragma unroll
    for (int i = 0; i < 4; ++i) {
        const int c = tid + i * 256;
        const float o = (v[i] - mu) * inv * g[c] + bt[c];
        y[(size_t)row * 1024 + c] = o;
        if (H16) y16[(size_t)row * 1024 + c] = (_Float16)o;
    }
}

// ---------------------------------------------------------------------------
extern "C" void kernel_launch(void* const* d_in, const int* in_sizes, int n_in,
                              void* d_out, int out_size, void* d_ws, size_t ws_size,
                              hipStream_t stream) {
    const float* src   = (const float*)d_in[0];
    const float* in_w  = (const float*)d_in[1];
    const float* in_b  = (const float*)d_in[2];
    const float* out_w = (const float*)d_in[3];
    const float* out_b = (const float*)d_in[4];
    const float* w1    = (const float*)d_in[5];
    const float* b1    = (const float*)d_in[6];
    const float* w2    = (const float*)d_in[7];
    const float* b2    = (const float*)d_in[8];
    const float* ln1g  = (const float*)d_in[9];
    const float* ln1b  = (const float*)d_in[10];
    const float* ln2g  = (const float*)d_in[11];
    const float* ln2b  = (const float*)d_in[12];
    float* out = (float*)d_out;

    constexpr int M = 4 * 8192;    // 32768 tokens
    char* ws = (char*)d_ws;
    // Workspace layout (bytes); FFN buffers overlay dead earlier regions.
    _Float16* qkv = (_Float16*)(ws);                  // 32768*3072 f16 = 192MB
    _Float16* ctx = (_Float16*)(ws + 201326592);      // 32768*1024 f16 =  64MB
    float*    x1  = (float*)   (ws + 268435456);      // src+attn f32    = 128MB
    float*    h   = (float*)   (ws + 402653184);      // LN1 out f32     = 128MB
    _Float16* h16 = (_Float16*)(ws + 536870912);      // LN1 out f16     =  64MB
    _Float16* ff1 = (_Float16*)(ws);                  // overlays qkv+ctx (256MB)
    float*    ff2 = (float*)   (ws + 268435456);      // overlays x1

    // 1. QKV projection: src(f32) x in_proj_w^T -> qkv (f16)
    gemm_kernel<false, false, false, false, true>
        <<<dim3(3072 / 128, M / 128), 256, GEMM_SMEM_BYTES, stream>>>(
            src, in_w, in_b, nullptr, nullptr, qkv, M, 3072, 1024);

    // 2. Block-local attention (4 batches x 64 windows x 16 heads)
    attn_kernel<<<dim3(4 * 64 * 16), 256, ATTN_SMEM_BYTES, stream>>>(qkv, ctx);

    // 3. Output projection + residual: x1 = src + ctx x out_proj_w^T
    gemm_kernel<true, false, true, true, false>
        <<<dim3(1024 / 128, M / 128), 256, GEMM_SMEM_BYTES, stream>>>(
            ctx, out_w, out_b, src, x1, nullptr, M, 1024, 1024);

    // 4. LayerNorm 1 -> h (f32) and h16 (f16)
    ln_kernel<true><<<dim3(M), 256, 0, stream>>>(x1, ln1g, ln1b, h, h16);

    // 5. FFN1: relu(h x w1^T + b1) -> ff1 (f16)
    gemm_kernel<true, true, false, false, true>
        <<<dim3(4096 / 128, M / 128), 256, GEMM_SMEM_BYTES, stream>>>(
            h16, w1, b1, nullptr, nullptr, ff1, M, 4096, 1024);

    // 6. FFN2 + residual: ff2 = h + ff1 x w2^T + b2
    gemm_kernel<true, false, true, true, false>
        <<<dim3(1024 / 128, M / 128), 256, GEMM_SMEM_BYTES, stream>>>(
            ff1, w2, b2, h, ff2, nullptr, M, 1024, 4096);

    // 7. LayerNorm 2 -> out
    ln_kernel<false><<<dim3(M), 256, 0, stream>>>(ff2, ln2g, ln2b, out, nullptr);
}
